// MultiheadSelfAttention_63213328662837
// MI455X (gfx1250) — matrix operand order repaired and checked by reading, not compiled, not run on hardware
//
#include <hip/hip_runtime.h>
#include <hip/hip_bf16.h>

typedef __bf16 bf16_t;
typedef __attribute__((ext_vector_type(16))) __bf16 v16bf;
typedef __attribute__((ext_vector_type(8)))  __bf16 v8bf;
typedef __attribute__((ext_vector_type(8)))  float  v8f;
typedef __attribute__((ext_vector_type(4)))  unsigned int u32x4;
typedef __attribute__((ext_vector_type(8)))  int i32x8;
typedef __attribute__((ext_vector_type(4)))  int i32x4;

constexpr int Bc  = 2;
constexpr int Tc  = 2048;
constexpr int Dc  = 1024;
constexpr int Hc  = 16;
constexpr int DHc = 64;
constexpr int Pc  = 512;
constexpr int Sc  = Pc + Tc;   // 2560
constexpr int BT  = Bc * Tc;   // 4096

// ---------------------------------------------------------------------------
// WMMA fragment helpers (this code's contiguous k order; A and B match)
// A 16x32 bf16: lanes0-15 M=lane, elems0-15 -> K=0..15 contiguous
//               lanes16-31 same M, K=16..31 (matches B order)
// B 32x16 bf16: lane half0 N=lane, elems -> K=0..15 contiguous; half1 K=16..31
// ---------------------------------------------------------------------------
__device__ __forceinline__ v16bf load_afrag(const bf16_t* row, int hf) {
  const v8bf lo = *(const v8bf*)(row + hf * 16);
  const v8bf hi = *(const v8bf*)(row + hf * 16 + 8);
  v16bf a;
#pragma unroll
  for (int i = 0; i < 8; ++i) { a[i] = lo[i]; a[i + 8] = hi[i]; }
  return a;
}

__device__ __forceinline__ v8f wmma_bf16(v16bf a, v16bf b, v8f c) {
  return __builtin_amdgcn_wmma_f32_16x16x32_bf16(false, a, false, b, (short)0, c,
                                                 false, false);
}

// ---------------------------------------------------------------------------
// TDM: DMA a 64(N-rows) x 32(K, contiguous) bf16 tile of W into LDS.
// D# group0: count=1 | lds_addr | global_addr[56:0] | type=2
// D# group1: data_size=2B; tensor_dim0=1024; tensor_dim1=1024; tile_dim0=32;
//            tile_dim1=64; tensor_dim0_stride=1024. Groups 2/3 zero (2-D).
// amdgpu-toolchain builtin arity: (u32x4, i32x8, i32x4, i32x4, i32x8, i32 cpol)
// ---------------------------------------------------------------------------
__device__ __forceinline__ void tdm_load_wtile(const bf16_t* W, int n0, int k0,
                                               unsigned int lds_addr) {
  unsigned long long ga = (unsigned long long)(uintptr_t)(W + (size_t)n0 * Dc + k0);
  u32x4 g0;
  g0[0] = 1u;                                   // count=1, user mode
  g0[1] = lds_addr;                             // LDS byte address
  g0[2] = (unsigned int)ga;                     // global addr [31:0]
  g0[3] = (unsigned int)(ga >> 32) | (2u << 30);// global addr [56:32] | type=2
  i32x8 g1;
  g1[0] = (int)(1u << 16);                      // data_size=1 (2 bytes)
  g1[1] = (int)((unsigned)Dc << 16);            // tensor_dim0[15:0]=1024
  g1[2] = (int)((unsigned)Dc << 16);            // tensor_dim1[15:0]=1024
  g1[3] = (int)(32u << 16);                     // tile_dim0=32
  g1[4] = 64;                                   // tile_dim1=64, tile_dim2=0
  g1[5] = Dc;                                   // tensor_dim0_stride=1024
  g1[6] = 0;
  g1[7] = 0;
  i32x4 z4 = {0, 0, 0, 0};
  i32x8 z8 = {0, 0, 0, 0, 0, 0, 0, 0};
  __builtin_amdgcn_tensor_load_to_lds(g0, g1, z4, z4, z8, 0);
}

// ---------------------------------------------------------------------------
// Elementwise conversions
// ---------------------------------------------------------------------------
__global__ void cvt_f32_bf16(const float* __restrict__ in, bf16_t* __restrict__ out,
                             int n) {
  int i = blockIdx.x * blockDim.x + threadIdx.x;
  if (i < n) out[i] = (bf16_t)in[i];
}

// prefix_k [B,H,P,DH] f32 -> kbuf [B,H,S,DH] rows 0..P-1
__global__ void cvt_prefix_k(const float* __restrict__ pk, bf16_t* __restrict__ kbuf) {
  int idx = blockIdx.x * blockDim.x + threadIdx.x;
  int dh = idx & 63;
  int p  = (idx >> 6) & (Pc - 1);
  int bh = idx >> 15;
  kbuf[((size_t)bh * Sc + p) * DHc + dh] = (bf16_t)pk[idx];
}

// prefix_v [B,H,P,DH] f32 -> vT [B,H,DH,S] cols 0..P-1
__global__ void cvt_prefix_v(const float* __restrict__ pv, bf16_t* __restrict__ vT) {
  int idx = blockIdx.x * blockDim.x + threadIdx.x;
  int dh = idx & 63;
  int p  = (idx >> 6) & (Pc - 1);
  int bh = idx >> 15;
  vT[((size_t)bh * DHc + dh) * Sc + p] = (bf16_t)pv[idx];
}

// ---------------------------------------------------------------------------
// GEMM: out = A[4096,1024] @ W.T, W row-major [N=1024, K=1024]
// Block = 4 waves; wave computes a 32(M) x 64(N) tile (2 A-frags, 8 accums).
// W tile (64x32) staged into LDS by the Tensor Data Mover, double-buffered,
// shared by all 4 waves. TENSORcnt + barriers for pipelining.
// MODE 0: bf16 plain [M,N]   (q)
// MODE 1: bf16 -> kbuf[B,H,S,DH] at row P+t   (k)
// MODE 2: bf16 -> vT[B,H,DH,S] at col P+t     (v, packed 8x stores)
// MODE 3: f32 plain [M,N]    (final output)
// ---------------------------------------------------------------------------
template <int MODE>
__global__ __launch_bounds__(128) void gemm_wmma(const bf16_t* __restrict__ A,
                                                 const bf16_t* __restrict__ W,
                                                 void* __restrict__ out) {
  constexpr int Kc = Dc, Nc = Dc;
  __shared__ __align__(64) bf16_t wtile[2][64][32];   // 2 x 4 KiB
  const int lane = threadIdx.x & 31;
  const int wave = threadIdx.x >> 5;
  const int hf   = lane >> 4;
  const int l16  = lane & 15;
  const int m0   = blockIdx.x * 128 + wave * 32;
  const int n0   = blockIdx.y * 64;

  const unsigned int lb0 = (unsigned int)(uintptr_t)&wtile[0][0][0];
  const unsigned int lb[2] = {lb0, lb0 + 64u * 32u * 2u};

  v8f acc[2][4] = {};
  const bf16_t* arow0 = A + (size_t)(m0 + l16) * Kc;
  const bf16_t* arow1 = A + (size_t)(m0 + 16 + l16) * Kc;

  if (wave == 0) tdm_load_wtile(W, n0, 0, lb[0]);

  int buf = 0;
  for (int k0 = 0; k0 < Kc; k0 += 32) {
    if (wave == 0) {
      if (k0 + 32 < Kc) {
        tdm_load_wtile(W, n0, k0 + 32, lb[buf ^ 1]);
        __builtin_amdgcn_s_wait_tensorcnt(1);
      } else {
        __builtin_amdgcn_s_wait_tensorcnt(0);
      }
    }
    __syncthreads();                       // tile[buf] visible to all waves
    v16bf af0 = load_afrag(arow0 + k0, hf);
    v16bf af1 = load_afrag(arow1 + k0, hf);
#pragma unroll
    for (int j = 0; j < 4; ++j) {
      v16bf wf = *(const v16bf*)(&wtile[buf][j * 16 + l16][hf * 16]);
      acc[0][j] = wmma_bf16(af0, wf, acc[0][j]);
      acc[1][j] = wmma_bf16(af1, wf, acc[1][j]);
    }
    __syncthreads();                       // everyone done before buf is re-filled
    buf ^= 1;
  }

  // C/D layout: vgpr r, lanes0-15 -> M=r, lanes16-31 -> M=8+r; N = n0+j*16+l16
#pragma unroll
  for (int ms = 0; ms < 2; ++ms) {
    const int mb = m0 + ms * 16;
    if constexpr (MODE == 0) {
      bf16_t* o = (bf16_t*)out;
#pragma unroll
      for (int j = 0; j < 4; ++j)
#pragma unroll
        for (int r = 0; r < 8; ++r)
          o[(size_t)(mb + hf * 8 + r) * Nc + n0 + j * 16 + l16] =
              (bf16_t)acc[ms][j][r];
    } else if constexpr (MODE == 1) {
      bf16_t* o = (bf16_t*)out;
#pragma unroll
      for (int j = 0; j < 4; ++j)
#pragma unroll
        for (int r = 0; r < 8; ++r) {
          int m = mb + hf * 8 + r;
          int b = m >> 11, t = m & (Tc - 1);
          int n = n0 + j * 16 + l16;
          int h = n >> 6, dh = n & 63;
          o[((size_t)(b * Hc + h) * Sc + Pc + t) * DHc + dh] = (bf16_t)acc[ms][j][r];
        }
    } else if constexpr (MODE == 2) {
      bf16_t* o = (bf16_t*)out;
      int b = mb >> 11;
      int t = (mb & (Tc - 1)) + hf * 8;
#pragma unroll
      for (int j = 0; j < 4; ++j) {
        int n = n0 + j * 16 + l16;
        int h = n >> 6, dh = n & 63;
        v8bf pack;
#pragma unroll
        for (int r = 0; r < 8; ++r) pack[r] = (bf16_t)acc[ms][j][r];
        *(v8bf*)(o + ((size_t)(b * Hc + h) * DHc + dh) * Sc + Pc + t) = pack;
      }
    } else {
      float* o = (float*)out;
#pragma unroll
      for (int j = 0; j < 4; ++j)
#pragma unroll
        for (int r = 0; r < 8; ++r)
          o[(size_t)(mb + hf * 8 + r) * Nc + n0 + j * 16 + l16] = acc[ms][j][r];
    }
  }
}

// ---------------------------------------------------------------------------
// Flash attention: one wave per (b, h, 16-row query tile).
// q:    [B*T, D] bf16        (A frags, dh-contiguous)
// kbuf: [B,H,S,DH] bf16      (score B frags, dh-contiguous)
// vT:   [B,H,DH,S] bf16      (PV B frags, s-contiguous)
// y:    [B*T, D] bf16
// ---------------------------------------------------------------------------
__global__ __launch_bounds__(128) void attn_flash(const bf16_t* __restrict__ q,
                                                  const bf16_t* __restrict__ kbuf,
                                                  const bf16_t* __restrict__ vT,
                                                  bf16_t* __restrict__ y) {
  __shared__ __align__(64) bf16_t lds_p[4][16][32];   // per-wave P-tile bounce
  const int lane = threadIdx.x & 31;
  const int wave = threadIdx.x >> 5;
  const int hf   = lane >> 4;
  const int l16  = lane & 15;
  const int mt   = blockIdx.x * 4 + wave;     // 0..127
  const int h    = blockIdx.y;
  const int b    = blockIdx.z;
  const int t0   = mt * 16;

  const bf16_t* qrow = q + (size_t)(b * Tc + t0 + l16) * Dc + h * DHc;
  v16bf aq0 = load_afrag(qrow, hf);        // K(=dh) 0..31
  v16bf aq1 = load_afrag(qrow + 32, hf);   // K(=dh) 32..63

  const bf16_t* kb = kbuf + (size_t)(b * Hc + h) * Sc * DHc;
  const bf16_t* vb = vT + (size_t)(b * Hc + h) * DHc * Sc;

  float rowmax[8], rowsum[8];
#pragma unroll
  for (int r = 0; r < 8; ++r) { rowmax[r] = -3.0e38f; rowsum[r] = 0.f; }
  v8f acc[4] = {};

  constexpr float LOG2E = 1.44269504088896f;
  constexpr float SCALE = 0.125f;             // 1/sqrt(64)

  for (int s0 = 0; s0 < Sc; s0 += 32) {
    // prefetch next chunk of K and V (global_prefetch_b8)
    if (s0 + 32 < Sc) {
      __builtin_prefetch(kb + (size_t)(s0 + 32 + l16) * DHc, 0, 0);
      __builtin_prefetch(vb + (size_t)(lane * 2) * Sc + s0 + 32, 0, 0);
    }
    // ---- scores: two 16x16 tiles, K=64 in two steps
    v8f sc[2] = {};
#pragma unroll
    for (int nt = 0; nt < 2; ++nt) {
      const bf16_t* kr = kb + (size_t)(s0 + nt * 16 + l16) * DHc + hf * 16;
      v16bf b0 = *(const v16bf*)(kr);
      v16bf b1 = *(const v16bf*)(kr + 32);
      sc[nt] = wmma_bf16(aq0, b0, sc[nt]);
      sc[nt] = wmma_bf16(aq1, b1, sc[nt]);
    }
    // ---- online softmax (row stats live in each 16-lane half)
    float pv0[8], pv1[8];
#pragma unroll
    for (int r = 0; r < 8; ++r) {
      float sa = sc[0][r] * SCALE;
      float sb = sc[1][r] * SCALE;
      float mx = fmaxf(sa, sb);
#pragma unroll
      for (int off = 1; off < 16; off <<= 1) mx = fmaxf(mx, __shfl_xor(mx, off, 32));
      float nm = fmaxf(rowmax[r], mx);
      float corr = exp2f((rowmax[r] - nm) * LOG2E);
      rowmax[r] = nm;
      float pa = exp2f((sa - nm) * LOG2E);
      float pb = exp2f((sb - nm) * LOG2E);
      pv0[r] = pa; pv1[r] = pb;
      float ps = pa + pb;
#pragma unroll
      for (int off = 1; off < 16; off <<= 1) ps += __shfl_xor(ps, off, 32);
      rowsum[r] = rowsum[r] * corr + ps;
#pragma unroll
      for (int j = 0; j < 4; ++j) acc[j][r] *= corr;
    }
    // ---- D-layout -> A-layout transpose of P through LDS
#pragma unroll
    for (int r = 0; r < 8; ++r) {
      lds_p[wave][hf * 8 + r][l16]      = (bf16_t)pv0[r];
      lds_p[wave][hf * 8 + r][16 + l16] = (bf16_t)pv1[r];
    }
    asm volatile("s_wait_dscnt 0" ::: "memory");
    v16bf ap = load_afrag(&lds_p[wave][l16][0], hf);
    // ---- O += P @ V   (4 dh tiles)
#pragma unroll
    for (int j = 0; j < 4; ++j) {
      const bf16_t* vr = vb + (size_t)(j * 16 + l16) * Sc + s0 + hf * 16;
      v16bf bv = *(const v16bf*)vr;
      acc[j] = wmma_bf16(ap, bv, acc[j]);
    }
  }

  // ---- normalize and write [b, t, h*64+dh]
  float inv[8];
#pragma unroll
  for (int r = 0; r < 8; ++r) inv[r] = 1.f / rowsum[r];
#pragma unroll
  for (int j = 0; j < 4; ++j)
#pragma unroll
    for (int r = 0; r < 8; ++r)
      y[(size_t)(b * Tc + t0 + hf * 8 + r) * Dc + h * DHc + j * 16 + l16] =
          (bf16_t)(acc[j][r] * inv[r]);
}

// ---------------------------------------------------------------------------
extern "C" void kernel_launch(void* const* d_in, const int* in_sizes, int n_in,
                              void* d_out, int out_size, void* d_ws, size_t ws_size,
                              hipStream_t stream) {
  (void)in_sizes; (void)n_in; (void)out_size; (void)ws_size;
  const float* x  = (const float*)d_in[0];
  const float* pk = (const float*)d_in[1];
  const float* pv = (const float*)d_in[2];
  const float* Wq = (const float*)d_in[3];
  const float* Wk = (const float*)d_in[4];
  const float* Wv = (const float*)d_in[5];
  const float* Wo = (const float*)d_in[6];

  char* ws = (char*)d_ws;
  size_t off = 0;
  auto carve = [&](size_t elems) {
    bf16_t* p = (bf16_t*)(ws + off);
    off += (elems * sizeof(bf16_t) + 255) & ~(size_t)255;
    return p;
  };
  bf16_t* xbf  = carve((size_t)BT * Dc);       // 8 MiB
  bf16_t* wqb  = carve((size_t)Dc * Dc);       // 2 MiB
  bf16_t* wkb  = carve((size_t)Dc * Dc);
  bf16_t* wvb  = carve((size_t)Dc * Dc);
  bf16_t* wob  = carve((size_t)Dc * Dc);
  bf16_t* qbuf = carve((size_t)BT * Dc);       // 8 MiB
  bf16_t* kbuf = carve((size_t)Bc * Hc * Sc * DHc);  // 10 MiB
  bf16_t* vTb  = carve((size_t)Bc * Hc * DHc * Sc);  // 10 MiB
  bf16_t* ybf  = carve((size_t)BT * Dc);       // 8 MiB  => ~52 MiB total

  const int thr = 256;
  cvt_f32_bf16<<<(BT * Dc) / thr, thr, 0, stream>>>(x, xbf, BT * Dc);
  cvt_f32_bf16<<<(Dc * Dc) / thr, thr, 0, stream>>>(Wq, wqb, Dc * Dc);
  cvt_f32_bf16<<<(Dc * Dc) / thr, thr, 0, stream>>>(Wk, wkb, Dc * Dc);
  cvt_f32_bf16<<<(Dc * Dc) / thr, thr, 0, stream>>>(Wv, wvb, Dc * Dc);
  cvt_f32_bf16<<<(Dc * Dc) / thr, thr, 0, stream>>>(Wo, wob, Dc * Dc);
  cvt_prefix_k<<<(Bc * Hc * Pc * DHc) / thr, thr, 0, stream>>>(pk, kbuf);
  cvt_prefix_v<<<(Bc * Hc * Pc * DHc) / thr, thr, 0, stream>>>(pv, vTb);

  dim3 ggrid(BT / 128, Dc / 64);
  gemm_wmma<0><<<ggrid, 128, 0, stream>>>(xbf, wqb, (void*)qbuf);
  gemm_wmma<1><<<ggrid, 128, 0, stream>>>(xbf, wkb, (void*)kbuf);
  gemm_wmma<2><<<ggrid, 128, 0, stream>>>(xbf, wvb, (void*)vTb);

  attn_flash<<<dim3(Tc / 16 / 4, Hc, Bc), 128, 0, stream>>>(qbuf, kbuf, vTb, ybf);

  gemm_wmma<3><<<ggrid, 128, 0, stream>>>(ybf, wob, d_out);
}
